// MultiHeadQuantumAttention_18854906429930
// MI455X (gfx1250) — matrix operand order repaired and checked by reading, not compiled, not run on hardware
//
#include <hip/hip_runtime.h>
#include <math.h>

// ---------------------------------------------------------------------------
// MI455X (gfx1250) fused multi-head attention forward.
// Matmuls on v_wmma_f32_16x16x32_f16 (wave32, f32 accum); attention K/V tiles
// staged with async global->LDS DMA (ASYNCcnt) and double buffering.
// ---------------------------------------------------------------------------

typedef __attribute__((ext_vector_type(16))) _Float16 v16h;
typedef __attribute__((ext_vector_type(8)))  float    v8f;
typedef int v4i __attribute__((vector_size(16)));   // 16B chunk for async DMA

#define HEADS    16
#define HEAD_DIM 64
#define DIM      1024
#define BATCH    2
#define SEQ      2048
#define ROWS     (BATCH*SEQ)   // 4096

#if defined(__AMDGCN__) && __has_builtin(__builtin_amdgcn_global_load_async_to_lds_b128)
#define HAVE_ASYNC_LDS 1
#else
#define HAVE_ASYNC_LDS 0
#endif

__device__ __forceinline__ v8f zero8() {
  v8f z = {0.f,0.f,0.f,0.f,0.f,0.f,0.f,0.f};
  return z;
}

__device__ __forceinline__ v8f wmma_f16(v16h a, v16h b, v8f c) {
  // D = A(16x32 f16) * B(32x16 f16) + C(16x16 f32)
  return __builtin_amdgcn_wmma_f32_16x16x32_f16(
      /*neg_a=*/false, a, /*neg_b=*/false, b,
      /*c_mod=*/(short)0, c, /*reuse_a=*/false, /*reuse_b=*/false);
}

template<int N>
__device__ __forceinline__ void wait_asynccnt_le() {
#if defined(__AMDGCN__)
#if __has_builtin(__builtin_amdgcn_s_wait_asynccnt)
  __builtin_amdgcn_s_wait_asynccnt(N);
#else
  asm volatile("s_wait_asynccnt %0" :: "i"(N));
#endif
#endif
}

// ---------------------------------------------------------------------------
// GEMM: C[M,N] = A[M,K] @ W[N,K]^T + bias   (torch Linear semantics)
// Tile: 128(M) x 64(N), K-step 32. 256 threads = 8 waves (4x2 of 32x32).
// mode 0: row-major [B*S, DIM]; mode 1: scatter to [B, H, S, D].
// ---------------------------------------------------------------------------
#define GT_M 128
#define GT_N 64
#define GT_K 32
#define A_STR 40
#define W_STR 40

__global__ __launch_bounds__(256)
void gemm_bias_kernel(const float* __restrict__ A, const float* __restrict__ W,
                      const float* __restrict__ bias, float* __restrict__ out,
                      int mode)
{
  __shared__ _Float16 Ah[GT_M * A_STR];
  __shared__ _Float16 Wh[GT_N * W_STR];

  const int tid  = threadIdx.x;
  const int lane = tid & 31;
  const int wid  = tid >> 5;
  const int wm   = wid & 3;
  const int wn   = wid >> 2;
  const int m0   = blockIdx.x * GT_M;
  const int n0   = blockIdx.y * GT_N;
  const int hx   = lane >> 4;
  const int l16  = lane & 15;

  v8f acc[2][2];
#pragma unroll
  for (int i = 0; i < 2; ++i)
#pragma unroll
    for (int j = 0; j < 2; ++j) acc[i][j] = zero8();

  for (int k0 = 0; k0 < DIM; k0 += GT_K) {
#pragma unroll
    for (int i = 0; i < 4; ++i) {
      int id = tid + i * 256;
      int r  = id >> 3;
      int c  = (id & 7) * 4;
      float4 f = *(const float4*)(A + (size_t)(m0 + r) * DIM + k0 + c);
      _Float16* p = &Ah[r * A_STR + c];
      p[0] = (_Float16)f.x; p[1] = (_Float16)f.y;
      p[2] = (_Float16)f.z; p[3] = (_Float16)f.w;
    }
#pragma unroll
    for (int i = 0; i < 2; ++i) {
      int id = tid + i * 256;
      int r  = id >> 3;
      int c  = (id & 7) * 4;
      float4 f = *(const float4*)(W + (size_t)(n0 + r) * DIM + k0 + c);
      _Float16* p = &Wh[r * W_STR + c];
      p[0] = (_Float16)f.x; p[1] = (_Float16)f.y;
      p[2] = (_Float16)f.z; p[3] = (_Float16)f.w;
    }
    __syncthreads();

    v16h af[2], bf[2];
#pragma unroll
    for (int i = 0; i < 2; ++i) {
      const _Float16* pa = &Ah[(wm * 32 + i * 16 + l16) * A_STR];
      const int kb = hx * 16;
      v16h a;
#pragma unroll
      for (int j = 0; j < 8; ++j) { a[j] = pa[kb + j]; a[8 + j] = pa[kb + 8 + j]; }
      af[i] = a;
    }
#pragma unroll
    for (int j = 0; j < 2; ++j) {
      const _Float16* pb = &Wh[(wn * 32 + j * 16 + l16) * W_STR + hx * 16];
      v16h b;
#pragma unroll
      for (int q = 0; q < 16; ++q) b[q] = pb[q];
      bf[j] = b;
    }
#pragma unroll
    for (int i = 0; i < 2; ++i)
#pragma unroll
      for (int j = 0; j < 2; ++j)
        acc[i][j] = wmma_f16(af[i], bf[j], acc[i][j]);

    __syncthreads();
  }

#pragma unroll
  for (int i = 0; i < 2; ++i)
#pragma unroll
    for (int j = 0; j < 2; ++j) {
#pragma unroll
      for (int e = 0; e < 8; ++e) {
        int gr = m0 + wm * 32 + i * 16 + e + 8 * hx;
        int gc = n0 + wn * 32 + j * 16 + l16;
        float v = acc[i][j][e] + bias[gc];
        if (mode == 0) {
          out[(size_t)gr * DIM + gc] = v;
        } else {
          int b = gr >> 11, s = gr & (SEQ - 1);
          int hh = gc >> 6, d = gc & (HEAD_DIM - 1);
          out[(((size_t)(b * HEADS + hh)) * SEQ + s) * HEAD_DIM + d] = v;
        }
      }
    }
}

// ---------------------------------------------------------------------------
// Fused RoPE + f16 pack over a [B,H,S,D] fp32 buffer.
//   mode 0: plain f16 pack (V)
//   mode 1: RoPE + 1/sqrt(HEAD_DIM) pre-scale, f16 only (Q)
//   mode 2: RoPE, write fp32 back in place AND f16 (K: fp32 copy is returned)
// One thread per (b,h,s,d<32) rotation pair.
// ---------------------------------------------------------------------------
__global__ __launch_bounds__(256)
void rope_pack_kernel(const float* __restrict__ in, float* __restrict__ f32out,
                      _Float16* __restrict__ hout, int mode)
{
  int idx = blockIdx.x * blockDim.x + threadIdx.x;
  int d  = idx & 31;
  int s  = (idx >> 5) & (SEQ - 1);
  int bh = idx >> 16;
  size_t base = ((size_t)bh * SEQ + s) * HEAD_DIM;
  float x1 = in[base + d];
  float x2 = in[base + d + 32];
  float y1 = x1, y2 = x2;
  if (mode != 0) {
    float inv = __expf(-(float)d * (9.210340371976184f / 32.0f)); // 10000^(-d/32)
    float ang = (float)s * inv;
    float c = cosf(ang), sn = sinf(ang);
    y1 = x1 * c - x2 * sn;
    y2 = x2 * c + x1 * sn;
  }
  if (mode == 2) { f32out[base + d] = y1; f32out[base + d + 32] = y2; }
  float sc = (mode == 1) ? 0.125f : 1.0f;
  hout[base + d]      = (_Float16)(y1 * sc);
  hout[base + d + 32] = (_Float16)(y2 * sc);
}

// ---------------------------------------------------------------------------
// Flash attention per (b, h, 64-row q tile). 128 threads = 4 waves x 16 rows.
// K/V tiles (f16, padded rows) staged via async global->LDS DMA with double
// buffering: issue tile i+1, s_wait_asynccnt<=8 for tile i, barrier, compute.
// ---------------------------------------------------------------------------
#define AT_Q   64
#define AT_K   64
#define KV_STR 72   // 64 + 8 pad halves; row stride 144 B (16B aligned)

__device__ __forceinline__ void copy_tile(const _Float16* __restrict__ g,
                                          _Float16* l, int tid)
{
#pragma unroll
  for (int i = 0; i < 4; ++i) {
    int id = tid + i * 128;          // 512 chunks of 8 halves (16 B)
    int r  = id >> 3;
    int c  = (id & 7) * 8;
#if HAVE_ASYNC_LDS
    __builtin_amdgcn_global_load_async_to_lds_b128(
        (__attribute__((address_space(1))) v4i*)(g + (size_t)r * HEAD_DIM + c),
        (__attribute__((address_space(3))) v4i*)(l + r * KV_STR + c),
        0, 0);
#else
    *(uint4*)(l + r * KV_STR + c) =
        *(const uint4*)(g + (size_t)r * HEAD_DIM + c);
#endif
  }
}

__global__ __launch_bounds__(128)
void attention_kernel(const _Float16* __restrict__ Qh,
                      const _Float16* __restrict__ Kg,
                      const _Float16* __restrict__ Vg,
                      const int* __restrict__ mask,
                      float* __restrict__ ctx)
{
  __shared__ _Float16 Kh[2][AT_K * KV_STR];
  __shared__ _Float16 Vh[2][AT_K * KV_STR];
  __shared__ _Float16 Ph[4 * 16 * KV_STR];

  const int tid  = threadIdx.x;
  const int lane = tid & 31;
  const int wid  = tid >> 5;
  const int hx   = lane >> 4;
  const int l16  = lane & 15;
  const int q0   = blockIdx.x * AT_Q;
  const int h    = blockIdx.y;
  const int b    = blockIdx.z;
  const int qrow0 = q0 + wid * 16;

  const size_t headbase = ((size_t)(b * HEADS + h)) * SEQ * HEAD_DIM;
  const _Float16* qptr  = Qh + headbase + (size_t)qrow0 * HEAD_DIM;
  const _Float16* kbase = Kg + headbase;
  const _Float16* vbase = Vg + headbase;

  // Q fragments (pre-scaled f16): 8-half contiguous gathers -> b128 loads
  v16h aq[2];
#pragma unroll
  for (int ks = 0; ks < 2; ++ks) {
    const int kb = hx * 16;
    v16h a;
#pragma unroll
    for (int j = 0; j < 8; ++j) {
      a[j]     = qptr[(size_t)l16 * HEAD_DIM + ks * 32 + kb + j];
      a[8 + j] = qptr[(size_t)l16 * HEAD_DIM + ks * 32 + kb + 8 + j];
    }
    aq[ks] = a;
  }

  v8f o[4];
#pragma unroll
  for (int dt = 0; dt < 4; ++dt) o[dt] = zero8();
  float mrow[8], lrow[8];
#pragma unroll
  for (int e = 0; e < 8; ++e) { mrow[e] = -3.0e38f; lrow[e] = 0.f; }

  // prologue: DMA tile 0 into buffer 0
  copy_tile(kbase, &Kh[0][0], tid);
  copy_tile(vbase, &Vh[0][0], tid);

  int buf = 0;
  for (int kb0 = 0; kb0 < SEQ; kb0 += AT_K) {
    const bool hasnext = (kb0 + AT_K) < SEQ;
    if (hasnext) {
      copy_tile(kbase + (size_t)(kb0 + AT_K) * HEAD_DIM, &Kh[buf ^ 1][0], tid);
      copy_tile(vbase + (size_t)(kb0 + AT_K) * HEAD_DIM, &Vh[buf ^ 1][0], tid);
      wait_asynccnt_le<8>();   // current tile's 8 issues have landed
      __builtin_prefetch(kbase + (size_t)(kb0 + 2 * AT_K) * HEAD_DIM + tid * 32, 0, 1);
    } else {
      wait_asynccnt_le<0>();
    }
    __syncthreads();

    const _Float16* Kt = &Kh[buf][0];
    const _Float16* Vt = &Vh[buf][0];

    // ---- S = Q K^T : 4 key tiles x 2 d-steps ----
    v8f sc[4];
#pragma unroll
    for (int t = 0; t < 4; ++t) {
      sc[t] = zero8();
#pragma unroll
      for (int ks = 0; ks < 2; ++ks) {
        const _Float16* pb = &Kt[(t * 16 + l16) * KV_STR + ks * 32 + hx * 16];
        v16h bk;
#pragma unroll
        for (int q = 0; q < 16; ++q) bk[q] = pb[q];
        sc[t] = wmma_f16(aq[ks], bk, sc[t]);
      }
    }

    // ---- mask + online softmax (rows span a 16-lane group) ----
    float pv[4][8];
#pragma unroll
    for (int e = 0; e < 8; ++e) {
      const int m    = e + 8 * hx;
      const int grow = qrow0 + m;
      float rmax = -3.0e38f;
#pragma unroll
      for (int t = 0; t < 4; ++t) {
        int kcol = kb0 + t * 16 + l16;
        int mk = mask[((size_t)b * SEQ + grow) * SEQ + kcol];
        float s = sc[t][e];
        if (mk == 0) s = -3.0e38f;
        pv[t][e] = s;
        rmax = fmaxf(rmax, s);
      }
#pragma unroll
      for (int off = 1; off < 16; off <<= 1)
        rmax = fmaxf(rmax, __shfl_xor(rmax, off, 32));
      float mnew  = fmaxf(mrow[e], rmax);
      float alpha = __expf(mrow[e] - mnew);
      float psum  = 0.f;
#pragma unroll
      for (int t = 0; t < 4; ++t) {
        float s = pv[t][e];
        float p = (s <= -1.0e37f) ? 0.f : __expf(s - mnew);
        pv[t][e] = p;
        psum += p;
      }
#pragma unroll
      for (int off = 1; off < 16; off <<= 1)
        psum += __shfl_xor(psum, off, 32);
      lrow[e] = lrow[e] * alpha + psum;
      mrow[e] = mnew;
#pragma unroll
      for (int dt = 0; dt < 4; ++dt) o[dt][e] *= alpha;
    }

    // ---- P: C layout -> A layout through wave-private LDS tile ----
#pragma unroll
    for (int t = 0; t < 4; ++t)
#pragma unroll
      for (int e = 0; e < 8; ++e) {
        int m = e + 8 * hx;
        Ph[(wid * 16 + m) * KV_STR + t * 16 + l16] = (_Float16)pv[t][e];
      }

    // ---- O += P V ----
#pragma unroll
    for (int ks = 0; ks < 2; ++ks) {
      const _Float16* pp = &Ph[(wid * 16 + l16) * KV_STR + ks * 32];
      const int kb = hx * 16;
      v16h ap;
#pragma unroll
      for (int j = 0; j < 8; ++j) { ap[j] = pp[kb + j]; ap[8 + j] = pp[kb + 8 + j]; }
#pragma unroll
      for (int dt = 0; dt < 4; ++dt) {
        v16h bv;
#pragma unroll
        for (int q = 0; q < 16; ++q)
          bv[q] = Vt[(ks * 32 + hx * 16 + q) * KV_STR + dt * 16 + l16];
        o[dt] = wmma_f16(ap, bv, o[dt]);
      }
    }

    __syncthreads();   // all waves done with buffer `buf` before it is re-DMA'd
    buf ^= 1;
  }

  // ---- normalize + store context as [B*S, DIM] ----
  float invr[8];
#pragma unroll
  for (int e = 0; e < 8; ++e) invr[e] = (lrow[e] > 0.f) ? (1.f / lrow[e]) : 0.f;
#pragma unroll
  for (int dt = 0; dt < 4; ++dt)
#pragma unroll
    for (int e = 0; e < 8; ++e) {
      int srow = qrow0 + e + 8 * hx;
      int col  = h * HEAD_DIM + dt * 16 + l16;
      ctx[((size_t)b * SEQ + srow) * DIM + col] = o[dt][e] * invr[e];
    }
}

// ---------------------------------------------------------------------------
// Launch: QKV proj -> RoPE+pack -> flash attention -> out projection.
// d_out = [ out (B,S,DIM) | k (B,H,S,D) post-RoPE fp32 | v (B,H,S,D) fp32 ]
// d_ws  = [ qf32/ctx alias (16MB) | qh f16 (8MB) | kh f16 (8MB) | vh f16 (8MB) ]
// ---------------------------------------------------------------------------
extern "C" void kernel_launch(void* const* d_in, const int* in_sizes, int n_in,
                              void* d_out, int out_size, void* d_ws, size_t ws_size,
                              hipStream_t stream)
{
  (void)in_sizes; (void)n_in; (void)out_size; (void)ws_size;
  const float* query = (const float*)d_in[0];
  const float* key   = (const float*)d_in[1];
  const float* value = (const float*)d_in[2];
  const int*   mask  = (const int*)  d_in[3];
  const float* Wq = (const float*)d_in[4];  const float* bq = (const float*)d_in[5];
  const float* Wk = (const float*)d_in[6];  const float* bk = (const float*)d_in[7];
  const float* Wv = (const float*)d_in[8];  const float* bv = (const float*)d_in[9];
  const float* Wo = (const float*)d_in[10]; const float* bo = (const float*)d_in[11];

  float* out  = (float*)d_out;                    // [B,S,DIM]
  float* kout = out  + (size_t)ROWS * DIM;        // [B,H,S,D] fp32 (returned)
  float* vout = kout + (size_t)ROWS * DIM;        // [B,H,S,D] fp32 (returned)

  float*     qf32 = (float*)d_ws;                 // [B,H,S,D] fp32 (pre-RoPE Q)
  float*     ctx  = qf32;                         // aliases qf32 (dead after pack)
  _Float16*  qh   = (_Float16*)(qf32 + (size_t)ROWS * DIM);
  _Float16*  kh   = qh + (size_t)ROWS * DIM;
  _Float16*  vh   = kh + (size_t)ROWS * DIM;

  dim3 gg(ROWS / GT_M, DIM / GT_N);               // (32, 16)
  gemm_bias_kernel<<<gg, 256, 0, stream>>>(query, Wq, bq, qf32, 1);
  gemm_bias_kernel<<<gg, 256, 0, stream>>>(key,   Wk, bk, kout, 1);
  gemm_bias_kernel<<<gg, 256, 0, stream>>>(value, Wv, bv, vout, 1);

  const int rt = BATCH * HEADS * SEQ * 32;        // one thread per rotation pair
  rope_pack_kernel<<<rt / 256, 256, 0, stream>>>(qf32, nullptr, qh, 1);
  rope_pack_kernel<<<rt / 256, 256, 0, stream>>>(kout, kout,    kh, 2);
  rope_pack_kernel<<<rt / 256, 256, 0, stream>>>(vout, nullptr, vh, 0);

  attention_kernel<<<dim3(SEQ / AT_Q, HEADS, BATCH), 128, 0, stream>>>(
      qh, kh, vh, mask, ctx);

  gemm_bias_kernel<<<gg, 256, 0, stream>>>(ctx, Wo, bo, out, 0);
}
